// SeparableCritic_82566451299052
// MI455X (gfx1250) — matrix operand order repaired and checked by reading, not compiled, not run on hardware
//
#include <hip/hip_runtime.h>
#include <hip/hip_bf16.h>
#include <math.h>
#include <stdint.h>

typedef __attribute__((ext_vector_type(16))) _Float16 v16h;
typedef __attribute__((ext_vector_type(8)))  float    v8f;

#define BDIM 4096
#define LATD 128

// Wave tile: (MI_T*16) x (NI_T*16) = 64 x 32
#define MI_T 4
#define NI_T 2

union Frag16 { v16h h; uint32_t u[8]; };

// ---------------------------------------------------------------------------
// f32 -> f16 elementwise convert
// ---------------------------------------------------------------------------
__global__ void __launch_bounds__(256)
cvt_f16_kernel(const float* __restrict__ src, _Float16* __restrict__ dst, int n) {
    int i = blockIdx.x * blockDim.x + threadIdx.x;
    if (i < n) dst[i] = (_Float16)src[i];
}

// ---------------------------------------------------------------------------
// Pack W [K x N] (row-major f32) into Wt [N x K] (row-major f16).
// Row-major-[N][K] is exactly the layout the B-fragment loads want
// (per-lane column n, contiguous K pairs).
// ---------------------------------------------------------------------------
__global__ void __launch_bounds__(256)
pack_wt_kernel(const float* __restrict__ W, _Float16* __restrict__ Wt, int K, int N) {
    int i = blockIdx.x * blockDim.x + threadIdx.x;
    int total = K * N;
    if (i < total) {
        int k = i / N;
        int n = i - k * N;
        Wt[(size_t)n * K + k] = (_Float16)W[i];
    }
}

// ---------------------------------------------------------------------------
// Fragment loaders (CDNA5 ISA 7.12.2, wave32):
//  A (16x32 f16): lane L -> row M = L%16; VGPR v holds K pair
//     k = (v/4)*16 + (L/16)*8 + (v%4)*2     (contiguous in row-major A)
//  B (32x16 f16): lane L -> col N = L%16; VGPR v holds K pair
//     k = (v/4)*16 + (L/16)*8 + (v%4)*2     (contiguous in row-major Bt)
//  D (16x16 f32): lane L -> col N = L%16; VGPR r -> row M = r + 8*(L/16)
// ---------------------------------------------------------------------------
__device__ __forceinline__ v16h load_a_frag(const uint32_t* Arow, int kb, int hsel) {
    Frag16 a;
#pragma unroll
    for (int v = 0; v < 8; ++v) {
        const int ak = kb + ((v >> 2) << 4) + (hsel << 3) + ((v & 3) << 1);
        a.u[v] = Arow[ak >> 1];
    }
    return a.h;
}

__device__ __forceinline__ v16h load_b_frag(const uint32_t* Brow, int kb, int hsel) {
    Frag16 b;
#pragma unroll
    for (int v = 0; v < 8; ++v) {
        const int bk = kb + ((v >> 2) << 4) + (hsel << 3) + ((v & 3) << 1);
        b.u[v] = Brow[bk >> 1];
    }
    return b.h;
}

// ---------------------------------------------------------------------------
// Register-blocked GEMM: C[M x N] = A[M x K] * Bt[N x K]^T (+bias) (ReLU?)
// One wave computes a 64x32 tile as a 4x2 grid of 16x16 WMMA tiles:
// each B (weight) fragment feeds 4 WMMAs, each A fragment feeds 2
// (AI ~= 21 FLOP/byte of L2 traffic), and the 8 independent accumulators
// let the scheduler overlap WMMA issue with the b128 fragment loads
// while dodging WMMA->WMMA same-accumulator hazards.
// ---------------------------------------------------------------------------
__global__ void __launch_bounds__(32)
gemm_wmma_kernel(const _Float16* __restrict__ A, const _Float16* __restrict__ Bt,
                 const float* __restrict__ bias,
                 float* __restrict__ Cf, _Float16* __restrict__ Ch,
                 int M, int N, int K, int relu) {
    const int lane = threadIdx.x & 31;
    const int hsel = lane >> 4;         // 0 or 1 (half-wave)
    const int l16  = lane & 15;
    const int m0 = blockIdx.y * (MI_T * 16);
    const int n0 = blockIdx.x * (NI_T * 16);

    const uint32_t* Arow[MI_T];
    const uint32_t* Brow[NI_T];
#pragma unroll
    for (int mi = 0; mi < MI_T; ++mi)
        Arow[mi] = (const uint32_t*)(A + (size_t)(m0 + mi * 16 + l16) * K);
#pragma unroll
    for (int ni = 0; ni < NI_T; ++ni)
        Brow[ni] = (const uint32_t*)(Bt + (size_t)(n0 + ni * 16 + l16) * K);

    v8f acc[MI_T][NI_T];
#pragma unroll
    for (int mi = 0; mi < MI_T; ++mi)
#pragma unroll
        for (int ni = 0; ni < NI_T; ++ni)
            acc[mi][ni] = (v8f){};

    for (int kb = 0; kb < K; kb += 32) {
        v16h a[MI_T], b[NI_T];
#pragma unroll
        for (int mi = 0; mi < MI_T; ++mi) a[mi] = load_a_frag(Arow[mi], kb, hsel);
#pragma unroll
        for (int ni = 0; ni < NI_T; ++ni) b[ni] = load_b_frag(Brow[ni], kb, hsel);
#pragma unroll
        for (int mi = 0; mi < MI_T; ++mi)
#pragma unroll
            for (int ni = 0; ni < NI_T; ++ni)
                acc[mi][ni] = __builtin_amdgcn_wmma_f32_16x16x32_f16(
                    false, a[mi], false, b[ni],
                    (short)0, acc[mi][ni], false, false);
    }

    float bb[NI_T];
#pragma unroll
    for (int ni = 0; ni < NI_T; ++ni)
        bb[ni] = bias ? bias[n0 + ni * 16 + l16] : 0.0f;

#pragma unroll
    for (int mi = 0; mi < MI_T; ++mi) {
#pragma unroll
        for (int r = 0; r < 8; ++r) {
            const int m = m0 + mi * 16 + r + (hsel << 3);
#pragma unroll
            for (int ni = 0; ni < NI_T; ++ni) {
                const int n = n0 + ni * 16 + l16;
                float v = acc[mi][ni][r] + bb[ni];
                if (relu) v = fmaxf(v, 0.0f);
                if (Cf) Cf[(size_t)m * N + n] = v;
                if (Ch) Ch[(size_t)m * N + n] = (_Float16)v;
            }
        }
    }
}

// ---------------------------------------------------------------------------
// L2-normalize each row of z [rows x 128] (f32), write f16 row-major.
// One wave per row: 32 lanes x 4 elements; shuffle reduction.
// ---------------------------------------------------------------------------
__global__ void __launch_bounds__(32)
l2norm_kernel(const float* __restrict__ z, _Float16* __restrict__ out) {
    const int row  = blockIdx.x;
    const int lane = threadIdx.x;
    const float* zr = z + (size_t)row * LATD;
    float v[4];
    float ss = 0.0f;
#pragma unroll
    for (int i = 0; i < 4; ++i) { v[i] = zr[lane + 32 * i]; ss += v[i] * v[i]; }
#pragma unroll
    for (int off = 16; off >= 1; off >>= 1) ss += __shfl_xor(ss, off, 32);
    const float inv = 1.0f / fmaxf(sqrtf(ss), 1e-12f);
    _Float16* orow = out + (size_t)row * LATD;
#pragma unroll
    for (int i = 0; i < 4; ++i) orow[lane + 32 * i] = (_Float16)(v[i] * inv);
}

// ---------------------------------------------------------------------------
// Per-row InfoNCE term: mi[i] = log(B) + s[i][i] - logsumexp(s[i][:])
// ---------------------------------------------------------------------------
__global__ void __launch_bounds__(256)
mi_kernel(const float* __restrict__ scores, float* __restrict__ mi) {
    __shared__ float red[256];
    const int row = blockIdx.x;
    const int t   = threadIdx.x;
    const float* s = scores + (size_t)row * BDIM;

    float mx = -INFINITY;
    for (int j = t; j < BDIM; j += 256) mx = fmaxf(mx, s[j]);
    red[t] = mx; __syncthreads();
    for (int o = 128; o >= 1; o >>= 1) {
        if (t < o) red[t] = fmaxf(red[t], red[t + o]);
        __syncthreads();
    }
    mx = red[0]; __syncthreads();

    float sum = 0.0f;
    for (int j = t; j < BDIM; j += 256) sum += __expf(s[j] - mx);
    red[t] = sum; __syncthreads();
    for (int o = 128; o >= 1; o >>= 1) {
        if (t < o) red[t] += red[t + o];
        __syncthreads();
    }
    if (t == 0) {
        const float lse = mx + logf(red[0]);
        mi[row] = logf((float)BDIM) + (s[row] - lse);
    }
}

// ---------------------------------------------------------------------------
// Host-side orchestration
// ---------------------------------------------------------------------------
extern "C" void kernel_launch(void* const* d_in, const int* in_sizes, int n_in,
                              void* d_out, int out_size, void* d_ws, size_t ws_size,
                              hipStream_t stream) {
    (void)in_sizes; (void)n_in; (void)out_size; (void)ws_size;

    const float* x   = (const float*)d_in[0];   // [4096 x 256]
    const float* y   = (const float*)d_in[1];   // [4096 x 64]
    const float* sW1 = (const float*)d_in[2];   // [256 x 1024]
    const float* sb1 = (const float*)d_in[3];
    const float* sW2 = (const float*)d_in[4];   // [1024 x 1024]
    const float* sb2 = (const float*)d_in[5];
    const float* sW3 = (const float*)d_in[6];   // [1024 x 128]
    const float* sb3 = (const float*)d_in[7];
    const float* kW1 = (const float*)d_in[8];   // [64 x 1024]
    const float* kb1 = (const float*)d_in[9];
    const float* kW2 = (const float*)d_in[10];  // [1024 x 1024]
    const float* kb2 = (const float*)d_in[11];
    const float* kW3 = (const float*)d_in[12];  // [1024 x 128]
    const float* kb3 = (const float*)d_in[13];

    float* scores = (float*)d_out;                       // [4096 x 4096]
    float* mi     = (float*)d_out + (size_t)BDIM * BDIM; // [4096]

    // Workspace layout (bytes); total ~29.6 MB
    char* ws = (char*)d_ws;
    size_t off = 0;
    _Float16* x16   = (_Float16*)(ws + off); off += (size_t)BDIM * 256  * 2;  // 2 MB
    _Float16* y16   = (_Float16*)(ws + off); off += (size_t)BDIM * 64   * 2;  // 0.5 MB
    _Float16* sW1t  = (_Float16*)(ws + off); off += (size_t)1024 * 256  * 2;
    _Float16* sW2t  = (_Float16*)(ws + off); off += (size_t)1024 * 1024 * 2;
    _Float16* sW3t  = (_Float16*)(ws + off); off += (size_t)128  * 1024 * 2;
    _Float16* kW1t  = (_Float16*)(ws + off); off += (size_t)1024 * 64   * 2;
    _Float16* kW2t  = (_Float16*)(ws + off); off += (size_t)1024 * 1024 * 2;
    _Float16* kW3t  = (_Float16*)(ws + off); off += (size_t)128  * 1024 * 2;
    _Float16* hA    = (_Float16*)(ws + off); off += (size_t)BDIM * 1024 * 2;  // 8 MB
    _Float16* hB    = (_Float16*)(ws + off); off += (size_t)BDIM * 1024 * 2;  // 8 MB
    float*    zx32  = (float*)   (ws + off); off += (size_t)BDIM * LATD * 4;
    float*    zy32  = (float*)   (ws + off); off += (size_t)BDIM * LATD * 4;
    _Float16* zx16  = (_Float16*)(ws + off); off += (size_t)BDIM * LATD * 2;
    _Float16* zy16  = (_Float16*)(ws + off); off += (size_t)BDIM * LATD * 2;

    // --- Stage 0: f16 conversion of activations ---
    {
        int n = BDIM * 256;
        cvt_f16_kernel<<<(n + 255) / 256, 256, 0, stream>>>(x, x16, n);
        n = BDIM * 64;
        cvt_f16_kernel<<<(n + 255) / 256, 256, 0, stream>>>(y, y16, n);
    }
    // --- Stage 0b: transpose-pack weights to f16 [N x K] ---
    {
        int n = 256 * 1024;
        pack_wt_kernel<<<(n + 255) / 256, 256, 0, stream>>>(sW1, sW1t, 256, 1024);
        n = 1024 * 1024;
        pack_wt_kernel<<<(n + 255) / 256, 256, 0, stream>>>(sW2, sW2t, 1024, 1024);
        n = 1024 * 128;
        pack_wt_kernel<<<(n + 255) / 256, 256, 0, stream>>>(sW3, sW3t, 1024, 128);
        n = 64 * 1024;
        pack_wt_kernel<<<(n + 255) / 256, 256, 0, stream>>>(kW1, kW1t, 64, 1024);
        n = 1024 * 1024;
        pack_wt_kernel<<<(n + 255) / 256, 256, 0, stream>>>(kW2, kW2t, 1024, 1024);
        n = 1024 * 128;
        pack_wt_kernel<<<(n + 255) / 256, 256, 0, stream>>>(kW3, kW3t, 1024, 128);
    }

    const int TM = MI_T * 16;  // 64 rows per wave tile
    const int TN = NI_T * 16;  // 32 cols per wave tile

    // --- State MLP: x -> hA -> hB -> zx32 ---
    gemm_wmma_kernel<<<dim3(1024 / TN, BDIM / TM), 32, 0, stream>>>(
        x16, sW1t, sb1, nullptr, hA, BDIM, 1024, 256, 1);
    gemm_wmma_kernel<<<dim3(1024 / TN, BDIM / TM), 32, 0, stream>>>(
        hA, sW2t, sb2, nullptr, hB, BDIM, 1024, 1024, 1);
    gemm_wmma_kernel<<<dim3(LATD / TN, BDIM / TM), 32, 0, stream>>>(
        hB, sW3t, sb3, zx32, nullptr, BDIM, LATD, 1024, 0);

    // --- Skill MLP: y -> hA -> hB -> zy32 (reuses ping-pong buffers) ---
    gemm_wmma_kernel<<<dim3(1024 / TN, BDIM / TM), 32, 0, stream>>>(
        y16, kW1t, kb1, nullptr, hA, BDIM, 1024, 64, 1);
    gemm_wmma_kernel<<<dim3(1024 / TN, BDIM / TM), 32, 0, stream>>>(
        hA, kW2t, kb2, nullptr, hB, BDIM, 1024, 1024, 1);
    gemm_wmma_kernel<<<dim3(LATD / TN, BDIM / TM), 32, 0, stream>>>(
        hB, kW3t, kb3, zy32, nullptr, BDIM, LATD, 1024, 0);

    // --- L2 normalize -> f16 ---
    l2norm_kernel<<<BDIM, 32, 0, stream>>>(zx32, zx16);
    l2norm_kernel<<<BDIM, 32, 0, stream>>>(zy32, zy16);

    // --- scores = zx @ zy^T (zy16 row-major [n][k] == B-operand layout) ---
    gemm_wmma_kernel<<<dim3(BDIM / TN, BDIM / TM), 32, 0, stream>>>(
        zx16, zy16, nullptr, scores, nullptr, BDIM, BDIM, LATD, 0);

    // --- mi = log(B) + diag(log_softmax(scores)) ---
    mi_kernel<<<BDIM, 256, 0, stream>>>(scores, mi);
}
